// HybridModel_23115513987348
// MI455X (gfx1250) — matrix operand order repaired and checked by reading, not compiled, not run on hardware
//
#include <hip/hip_runtime.h>
#include <hip/hip_bf16.h>

// MI455X / gfx1250:
//   conv+BN+ReLU+pool fused, implicit-GEMM on v_wmma_f32_16x16x32_f16 (wave32),
//   fp16 intermediates, LDS tiles staged via GLOBAL_LOAD_ASYNC_TO_LDS (when the
//   builtin exists) + pre-swizzled B fragments, compile-time-folded im2col
//   offsets, tiny quantum head as one wave per image.

#if defined(__has_builtin)
#if __has_builtin(__builtin_amdgcn_global_load_async_to_lds_b32) && \
    __has_builtin(__builtin_amdgcn_s_wait_asynccnt)
#define HAVE_ASYNC_LDS 1
#endif
#endif
#ifndef HAVE_ASYNC_LDS
#define HAVE_ASYNC_LDS 0
#endif

typedef __attribute__((ext_vector_type(16))) _Float16 v16h;
typedef __attribute__((ext_vector_type(8)))  float    v8f;

typedef __attribute__((address_space(1))) int* gint_p;
typedef __attribute__((address_space(3))) int* lint_p;

union AFrag { unsigned u[8]; _Float16 h[16]; v16h v; };
union H2U   { _Float16 h; unsigned short u; };

// LDS offset of im2col element K (compile-time foldable once K is a constant)
__device__ __forceinline__ constexpr int aoff(int K, int ROWS, int SW) {
    return ((K / 9) * ROWS + (K % 9) / 3) * SW + (K % 9) % 3;
}

// ---------------------------------------------------------------------------
// Fused conv3x3(SAME) + bias + BN + ReLU + 2x2 maxpool.
// GEMM view: M = 16 pixels (8x2 strip), N = 16 out-channels, K = CIN*9 (padded).
// One wave = one strip; block = 8 waves = STRIPSX strips x ROWPAIRS row-pairs.
// ---------------------------------------------------------------------------
template<int CIN, int COUT, int W, int ROWPAIRS, int STRIPSX, typename TIN>
__global__ __launch_bounds__(256)
void conv_bn_relu_pool(const TIN* __restrict__ in,
                       const float* __restrict__ wgt,   // [COUT][CIN][3][3]
                       const float* __restrict__ cbias,
                       const float* __restrict__ bng, const float* __restrict__ bnb,
                       const float* __restrict__ bnm, const float* __restrict__ bnv,
                       _Float16* __restrict__ out)      // [N][COUT][W/2][W/2] f16
{
    constexpr int KDIM = CIN * 9;
    constexpr int KT   = (KDIM + 31) / 32;   // K tiles of 32
    constexpr int COT  = COUT / 16;          // N tiles of 16
    constexpr int ROWS = 2 * ROWPAIRS + 2;   // staged rows (with halo)
    constexpr int SW   = W + 4;              // staged width: 2-col halo each side
    constexpr int WO   = W / 2;              // pooled output width

    alignas(16) __shared__ _Float16 s_in[CIN * ROWS * SW];
    __shared__ unsigned s_w[COT * KT * 256];   // B fragments, register-layout order

    const int tid   = threadIdx.x;
    const int n     = blockIdx.x;
    const int ybase = blockIdx.y * (2 * ROWPAIRS);

    // CDNA5 prefetch of the input tile start (global_prefetch_b8)
    __builtin_prefetch(&in[((size_t)(n * CIN) * W + (ybase > 0 ? ybase - 1 : 0)) * W], 0, 0);

    // ---- stage weights pre-swizzled into B register layout -----------------
    // B (32x16 KxN f16): VGPR v: K = (v/4)*16 + (hi?8:0) + (v%4)*2 (+e); N=lane&15
    for (int t = tid; t < COT * KT * 256; t += 256) {
        int v    = t & 7;
        int lane = (t >> 3) & 31;
        int kt   = (t >> 8) % KT;
        int ct   = (t >> 8) / KT;
        int co   = ct * 16 + (lane & 15);
        int k0   = kt * 32 + (v >> 2) * 16 + ((lane >= 16) ? 8 : 0) + (v & 3) * 2;
        unsigned pak = 0;
        #pragma unroll
        for (int e = 0; e < 2; ++e) {
            int K = k0 + e;
            float wv = 0.f;
            if (K < KDIM) {
                int ci = K / 9, r9 = K % 9;
                wv = wgt[((co * CIN + ci) * 3 + (r9 / 3)) * 3 + (r9 % 3)];
            }
            H2U c; c.h = (_Float16)wv;
            pak |= ((unsigned)c.u) << (16 * e);
        }
        s_w[t] = pak;
    }

    // ---- stage zero-padded input tile as f16 -------------------------------
    // staged col = x_global + 2 (pairs are 4B-aligned in both global and LDS)
    bool staged = false;
#if HAVE_ASYNC_LDS
    if constexpr (sizeof(TIN) == 2) {
        // fp16 input: raw async copy global->LDS of aligned b32 pairs,
        // DS zero-stores only for OOB pairs (disjoint, no ordering hazard).
        unsigned* sp = (unsigned*)s_in;
        constexpr int PR = SW / 2;                 // pairs per row
        for (int p = tid; p < CIN * ROWS * PR; p += 256) {
            int ci  = p / (ROWS * PR);
            int rem = p % (ROWS * PR);
            int r   = rem / PR;
            int c2  = rem % PR;
            int y   = ybase - 1 + r;
            int x0  = 2 * c2 - 2;
            if (y >= 0 && y < W && x0 >= 0 && x0 < W) {
                __builtin_amdgcn_global_load_async_to_lds_b32(
                    (gint_p)(in + ((size_t)(n * CIN + ci) * W + y) * W + x0),
                    (lint_p)(sp + p),
                    0, 0);
            } else {
                sp[p] = 0u;
            }
        }
        __builtin_amdgcn_s_wait_asynccnt(0);
        staged = true;
    }
#endif
    if (!staged) {
        for (int t = tid; t < CIN * ROWS * SW; t += 256) {
            int ci  = t / (ROWS * SW);
            int rem = t % (ROWS * SW);
            int r   = rem / SW;
            int col = rem % SW;
            int y = ybase - 1 + r;
            int x = col - 2;
            float v = 0.f;
            if (y >= 0 && y < W && x >= 0 && x < W)
                v = (float)in[((size_t)(n * CIN + ci) * W + y) * W + x];
            s_in[t] = (_Float16)v;
        }
    }
    __syncthreads();

    const int lane = tid & 31;
    const int wave = tid >> 5;
    const int sx   = wave % STRIPSX;        // strip x-position (8 px wide)
    const int rp   = wave / STRIPSX;        // row-pair within block
    const int M    = lane & 15;             // pixel within strip
    const int dx   = M & 7;
    const int dy   = M >> 3;
    const bool hiK = (lane >= 16);          // A layout: upper half holds K+8
    const int base = (rp * 2 + dy) * SW + sx * 8 + dx + 1;  // runtime part of gather addr

    v8f acc[COT];
    #pragma unroll
    for (int ct = 0; ct < COT; ++ct) {
        v8f z;
        #pragma unroll
        for (int e = 0; e < 8; ++e) z[e] = 0.f;
        acc[ct] = z;
    }

    // ---- implicit-GEMM main loop (fully unrolled: im2col offsets fold) -----
    #pragma unroll
    for (int kt = 0; kt < KT; ++kt) {
        // A (16x32 MxK f16): VGPR v: K = kt*32 + (v/4)*16 + (hi?8:0) + (v%4)*2 (+e)
        AFrag a;
        #pragma unroll
        for (int v = 0; v < 8; ++v) {
            #pragma unroll
            for (int e = 0; e < 2; ++e) {
                const int Klo = kt * 32 + (v >> 2) * 16 + (v & 3) * 2 + e;
                const int Khi = Klo + 8;
                const int offLo = (Klo < KDIM) ? aoff(Klo, ROWS, SW) : 0;
                const int offHi = (Khi < KDIM) ? aoff(Khi, ROWS, SW) : 0;
                const bool okLo = Klo < KDIM, okHi = Khi < KDIM;
                int  off = hiK ? offHi : offLo;   // cndmask of two immediates
                bool ok  = hiK ? okHi  : okLo;
                _Float16 ld = s_in[base + off];
                a.h[2 * v + e] = ok ? ld : (_Float16)0.f;
            }
        }
        #pragma unroll
        for (int ct = 0; ct < COT; ++ct) {
            AFrag b;
            #pragma unroll
            for (int v = 0; v < 8; ++v)
                b.u[v] = s_w[((ct * KT + kt) * 32 + lane) * 8 + v];
            acc[ct] = __builtin_amdgcn_wmma_f32_16x16x32_f16(
                false, a.v, false, b.v, (short)0, acc[ct], false, false);
        }
    }

    // ---- epilogue: bias+BN+ReLU, 2x2 maxpool, f16 store --------------------
    // D layout: VGPR r holds M=r (lanes 0-15) / M=r+8 (lanes 16-31), N=lane&15.
    // pool-x: max(VGPR 2k, 2k+1); pool-y: shfl_xor 16 (dy=0 vs dy=1 halves).
    const int py = blockIdx.y * ROWPAIRS + rp;
    #pragma unroll
    for (int ct = 0; ct < COT; ++ct) {
        int co = ct * 16 + (lane & 15);
        float inv  = bng[co] * rsqrtf(bnv[co] + 1e-5f);
        float beta = cbias[co] * inv + bnb[co] - bnm[co] * inv;
        #pragma unroll
        for (int k = 0; k < 4; ++k) {
            float e0 = fmaxf(acc[ct][2 * k]     * inv + beta, 0.f);
            float e1 = fmaxf(acc[ct][2 * k + 1] * inv + beta, 0.f);
            float p  = fmaxf(e0, e1);
            p = fmaxf(p, __shfl_xor(p, 16, 32));
            if (lane < 16)
                out[((size_t)(n * COUT + co) * WO + py) * WO + (sx * 4 + k)] = (_Float16)p;
        }
    }
}

// ---------------------------------------------------------------------------
// Head: proj (4096->4), 4-qubit quantum feature map (16 amplitudes, 4 CNOTs,
// probs @ FEAT -> 10), classifier 10x10. One wave per image.
// ---------------------------------------------------------------------------
__global__ __launch_bounds__(256)
void head_kernel(const _Float16* __restrict__ f,   // [1024][4096]
                 const float* __restrict__ pw,     // [4][4096]
                 const float* __restrict__ pb,     // [4]
                 const float* __restrict__ cw,     // [10][10]
                 const float* __restrict__ cb,     // [10]
                 float* __restrict__ outp)         // [1024][10]
{
    const int lane = threadIdx.x & 31;
    const int wave = threadIdx.x >> 5;
    const int n    = blockIdx.x * 8 + wave;

    float z[4] = {0.f, 0.f, 0.f, 0.f};
    const _Float16* fr = f + (size_t)n * 4096;
    for (int i = lane; i < 4096; i += 32) {
        float fv = (float)fr[i];
        #pragma unroll
        for (int j = 0; j < 4; ++j) z[j] += fv * pw[j * 4096 + i];
    }
    #pragma unroll
    for (int j = 0; j < 4; ++j) {
        #pragma unroll
        for (int off = 16; off > 0; off >>= 1)
            z[j] += __shfl_xor(z[j], off, 32);
        z[j] += pb[j];
    }

    // quantum feature map (computed redundantly per lane; lanes 0..9 store)
    float c4[4], s4[4];
    #pragma unroll
    for (int t = 0; t < 4; ++t) {
        float h = 0.5f * 1.57f * z[t];
        c4[t] = cosf(h); s4[t] = sinf(h);
    }
    float st[16];
    for (int m = 0; m < 16; ++m) {
        float p = 1.f;
        for (int t = 0; t < 4; ++t)
            p *= ((m >> (3 - t)) & 1) ? s4[t] : c4[t];
        st[m] = p;
    }
    const int cq[4] = {0, 1, 2, 3};
    const int tq[4] = {1, 2, 3, 0};
    for (int g = 0; g < 4; ++g) {
        float ns[16];
        for (int m = 0; m < 16; ++m) {
            int src = ((m >> (3 - cq[g])) & 1) ? (m ^ (1 << (3 - tq[g]))) : m;
            ns[m] = st[src];
        }
        for (int m = 0; m < 16; ++m) st[m] = ns[m];
    }
    const int pi[6] = {0, 0, 0, 1, 1, 2};
    const int pj[6] = {1, 2, 3, 2, 3, 3};
    float feat[10];
    for (int c = 0; c < 10; ++c) {
        float a = 0.f;
        for (int m = 0; m < 16; ++m) {
            float pr = st[m] * st[m];
            float sgn;
            if (c < 4) {
                sgn = 1.f - 2.f * (float)((m >> (3 - c)) & 1);
            } else {
                float si = 1.f - 2.f * (float)((m >> (3 - pi[c - 4])) & 1);
                float sj = 1.f - 2.f * (float)((m >> (3 - pj[c - 4])) & 1);
                sgn = si * sj;
            }
            a += pr * sgn;
        }
        feat[c] = a;
    }
    if (lane < 10) {
        float o = cb[lane];
        for (int j = 0; j < 10; ++j) o += feat[j] * cw[lane * 10 + j];
        outp[(size_t)n * 10 + lane] = o;
    }
}

// ---------------------------------------------------------------------------
extern "C" void kernel_launch(void* const* d_in, const int* in_sizes, int n_in,
                              void* d_out, int out_size, void* d_ws, size_t ws_size,
                              hipStream_t stream)
{
    (void)in_sizes; (void)n_in; (void)out_size; (void)ws_size;

    const float* x   = (const float*)d_in[0];
    const float* w1  = (const float*)d_in[1];
    const float* b1  = (const float*)d_in[2];
    const float* g1  = (const float*)d_in[3];
    const float* bb1 = (const float*)d_in[4];
    const float* m1  = (const float*)d_in[5];
    const float* v1  = (const float*)d_in[6];
    const float* w2  = (const float*)d_in[7];
    const float* b2  = (const float*)d_in[8];
    const float* g2  = (const float*)d_in[9];
    const float* bb2 = (const float*)d_in[10];
    const float* m2  = (const float*)d_in[11];
    const float* v2  = (const float*)d_in[12];
    const float* w3  = (const float*)d_in[13];
    const float* b3  = (const float*)d_in[14];
    const float* g3  = (const float*)d_in[15];
    const float* bb3 = (const float*)d_in[16];
    const float* m3  = (const float*)d_in[17];
    const float* v3  = (const float*)d_in[18];
    const float* pw  = (const float*)d_in[19];
    const float* pb  = (const float*)d_in[20];
    const float* cw  = (const float*)d_in[21];
    const float* cb  = (const float*)d_in[22];

    // fp16 intermediates in workspace (halves HBM traffic vs fp32)
    char* ws = (char*)d_ws;
    const size_t sz1 = (size_t)1024 * 16 * 32 * 32 * sizeof(_Float16); // 33.5 MB
    const size_t sz2 = (size_t)1024 * 32 * 16 * 16 * sizeof(_Float16); // 16.8 MB
    _Float16* o1 = (_Float16*)ws;
    _Float16* o2 = (_Float16*)(ws + sz1);
    _Float16* o3 = (_Float16*)(ws + sz1 + sz2);

    // conv1: 3->16, 64x64, 8 strips/row-pair, 1 row-pair/block
    conv_bn_relu_pool<3, 16, 64, 1, 8, float>
        <<<dim3(1024, 32), 256, 0, stream>>>(x, w1, b1, g1, bb1, m1, v1, o1);
    // conv2: 16->32, 32x32, 4 strips/row-pair, 2 row-pairs/block
    conv_bn_relu_pool<16, 32, 32, 2, 4, _Float16>
        <<<dim3(1024, 8), 256, 0, stream>>>(o1, w2, b2, g2, bb2, m2, v2, o2);
    // conv3: 32->64, 16x16, 2 strips/row-pair, 4 row-pairs/block
    conv_bn_relu_pool<32, 64, 16, 4, 2, _Float16>
        <<<dim3(1024, 2), 256, 0, stream>>>(o2, w3, b3, g3, bb3, m3, v3, o3);
    // head: 8 images per 256-thread block
    head_kernel<<<128, 256, 0, stream>>>(o3, pw, pb, cw, cb, (float*)d_out);
}
